// DBMG_35399120453814
// MI455X (gfx1250) — matrix operand order repaired and checked by reading, not compiled, not run on hardware
//
#include <hip/hip_runtime.h>

// ---------------------------------------------------------------------------
// Problem constants (from reference): E_IN=512, E_OUT=96, B=64, QL=128, PL=196
// ---------------------------------------------------------------------------

typedef _Float16 v16h __attribute__((ext_vector_type(16)));
typedef _Float16 v8h  __attribute__((ext_vector_type(8)));
typedef _Float16 v4h  __attribute__((ext_vector_type(4)));
typedef float    v8f  __attribute__((ext_vector_type(8)));

// --- CDNA5 async global->LDS staging (ASYNCcnt path), with safe fallback ---
#if defined(__gfx1250__) && __has_builtin(__builtin_amdgcn_global_load_async_to_lds_b128)
#define HAVE_ASYNC_LDS 1
typedef int i4 __attribute__((__vector_size__(4 * sizeof(int))));
typedef __attribute__((address_space(1))) i4* gptr_i4;
typedef __attribute__((address_space(3))) i4* lptr_i4;
__device__ __forceinline__ void async_copy_b128(const void* g, void* l) {
  __builtin_amdgcn_global_load_async_to_lds_b128((gptr_i4)g, (lptr_i4)l, 0, 0);
}
__device__ __forceinline__ void async_wait0() {
#if __has_builtin(__builtin_amdgcn_s_wait_asynccnt)
  __builtin_amdgcn_s_wait_asynccnt(0);
#else
  asm volatile("s_wait_asynccnt 0x0" ::: "memory");
#endif
}
#else
#define HAVE_ASYNC_LDS 0
#endif

// --- WMMA fragment helpers (contiguous k order for A and B, not AMD's table order) ---------
// A (16xK, 16-bit): lane l holds row M=l&15; halves at K = k0+16*(l>>4)..+15
//                   contiguous -> two 16-byte LDS reads.
// B (Kx16, 16-bit): lane l holds column N=l&15; halves at K = k0+16*(l>>4)..+15
//                   contiguous -> source stored as [n][k] row-major.
__device__ __forceinline__ v16h ld_2x8(const _Float16* p0, const _Float16* p1) {
  v8h a = *(const v8h*)p0;
  v8h b = *(const v8h*)p1;
  v16h r;
#pragma unroll
  for (int i = 0; i < 8; ++i) { r[i] = a[i]; r[i + 8] = b[i]; }
  return r;
}
__device__ __forceinline__ v16h loadA(const _Float16* base, int ld, int m0, int k0, int lane) {
  const _Float16* p = base + (size_t)(m0 + (lane & 15)) * ld + k0 + 16 * (lane >> 4);
  return ld_2x8(p, p + 8);
}
__device__ __forceinline__ v16h loadB(const _Float16* base, int ld, int n0, int k0, int lane) {
  const _Float16* p = base + (size_t)(n0 + (lane & 15)) * ld + k0 + 16 * (lane >> 4);
  return ld_2x8(p, p + 8);
}
__device__ __forceinline__ v8f wmma16(v16h a, v16h b, v8f c) {
  return __builtin_amdgcn_wmma_f32_16x16x32_f16(false, a, false, b, (short)0, c, false, false);
}

// ---------------------------------------------------------------------------
// Kernel 1: generic projection  Y[r][n] = sum_k X[r][k] * W[n][k] + b[n]
//   128 rows per workgroup, 8 waves -> wave w owns row-tile w (16 rows x 96).
//   K = 512 in 16 chunks of 32 staged through LDS as f16 (float4 loads).
//   Optional fused L2 row-normalization; f16 output (optionally transposed
//   [row/128][96][128] for the V tensor) and/or f32 output.
// ---------------------------------------------------------------------------
__global__ __launch_bounds__(256)
void proj_kernel(const float* __restrict__ X, int nrows,
                 const float* __restrict__ W, const float* __restrict__ bias,
                 int do_l2, int transpose,
                 _Float16* __restrict__ out_h, float* __restrict__ out_f) {
  __shared__ _Float16 Xs[128 * 32];
  __shared__ _Float16 Ws[96 * 32];
  __shared__ float    bs[96];
  const int tid  = threadIdx.x;
  const int lane = tid & 31;
  const int w    = tid >> 5;
  const int row0 = blockIdx.x * 128;

  if (tid < 96) bs[tid] = bias[tid];

  v8f acc[6];
#pragma unroll
  for (int t = 0; t < 6; ++t) acc[t] = v8f{};

  for (int ch = 0; ch < 16; ++ch) {
    __syncthreads();
    // X chunk: 128 rows x 32 cols as 1024 float4 loads -> packed f16 stores
    for (int i = tid; i < 1024; i += 256) {
      int r = i >> 3, c4 = (i & 7) * 4;
      int g = row0 + r;
      float4 v = make_float4(0.f, 0.f, 0.f, 0.f);
      if (g < nrows) v = *(const float4*)(X + (size_t)g * 512 + ch * 32 + c4);
      v4h h;
      h[0] = (_Float16)v.x; h[1] = (_Float16)v.y;
      h[2] = (_Float16)v.z; h[3] = (_Float16)v.w;
      *(v4h*)(Xs + r * 32 + c4) = h;
    }
    // W chunk: 96 rows x 32 cols as 768 float4 loads
    for (int i = tid; i < 768; i += 256) {
      int n = i >> 3, c4 = (i & 7) * 4;
      float4 v = *(const float4*)(W + (size_t)n * 512 + ch * 32 + c4);
      v4h h;
      h[0] = (_Float16)v.x; h[1] = (_Float16)v.y;
      h[2] = (_Float16)v.z; h[3] = (_Float16)v.w;
      *(v4h*)(Ws + n * 32 + c4) = h;
    }
    __syncthreads();
    v16h a = loadA(Xs, 32, w * 16, 0, lane);
#pragma unroll
    for (int t = 0; t < 6; ++t) {
      v16h b = loadB(Ws, 32, t * 16, 0, lane);
      acc[t] = wmma16(a, b, acc[t]);
    }
  }

  // Epilogue: bias, optional L2 norm, store.
  const int nb = lane & 15;
#pragma unroll
  for (int t = 0; t < 6; ++t) {
    float bv = bs[t * 16 + nb];
#pragma unroll
    for (int r = 0; r < 8; ++r) acc[t][r] += bv;
  }
  float inv[8];
#pragma unroll
  for (int r = 0; r < 8; ++r) inv[r] = 1.f;
  if (do_l2) {
#pragma unroll
    for (int r = 0; r < 8; ++r) {
      float ss = 0.f;
#pragma unroll
      for (int t = 0; t < 6; ++t) ss += acc[t][r] * acc[t][r];
      // lanes 0-15 hold row r, lanes 16-31 hold row r+8: reduce within halves
      for (int off = 8; off >= 1; off >>= 1) ss += __shfl_xor(ss, off, 16);
      inv[r] = 1.f / fmaxf(sqrtf(ss), 1e-6f);
    }
  }
#pragma unroll
  for (int r = 0; r < 8; ++r) {
    int g = row0 + w * 16 + ((lane < 16) ? r : r + 8);
    if (g >= nrows) continue;
#pragma unroll
    for (int t = 0; t < 6; ++t) {
      int n = t * 16 + nb;
      float v = acc[t][r] * inv[r];
      if (out_h) {
        size_t oidx = transpose
          ? ((size_t)(g >> 7) * (96 * 128) + (size_t)n * 128 + (g & 127))
          : ((size_t)g * 96 + n);
        out_h[oidx] = (_Float16)v;
      }
      if (out_f) out_f[(size_t)g * 96 + n] = v;
    }
  }
}

// ---------------------------------------------------------------------------
// Kernel 2: per-(b,c) cross attention -> S_TGL[b][c]
//   scores = Q[b] K[c]^T / sqrt(96); softmax over k; ctx = attn V[c];
//   mean over q; layer-norm; l2-norm; dot with d[b].
//   LDS (dynamic): Qs 24K | Ks 24K | Vt 24K | attn 32K | ctx 384B = 106880 B
// ---------------------------------------------------------------------------
#define ATTN_LDS_BYTES (24576 * 3 + 32768 + 96 * 4)
__global__ __launch_bounds__(256)
void attn_kernel(const _Float16* __restrict__ Qh,
                 const _Float16* __restrict__ Kh,
                 const _Float16* __restrict__ Vt,
                 const float* __restrict__ Pd,
                 const float* __restrict__ ln_g,
                 const float* __restrict__ ln_b,
                 float* __restrict__ S_TGL) {
  extern __shared__ char smem[];
  _Float16* Qs  = (_Float16*)(smem);
  _Float16* Ks  = (_Float16*)(smem + 24576);
  _Float16* Vs  = (_Float16*)(smem + 49152);          // [96][128] (V transposed)
  _Float16* As  = (_Float16*)(smem + 73728);          // [128][128] attn (f16)
  float*    ctx = (float*)(smem + 106496);            // [96]

  const int b = blockIdx.x, c = blockIdx.y;
  const int tid = threadIdx.x, lane = tid & 31, w = tid >> 5;

  { // stage operands: 12288 halves = 1536 b128 transfers each
    const v8h* s0 = (const v8h*)(Qh + (size_t)b * 12288);
    const v8h* s1 = (const v8h*)(Kh + (size_t)c * 12288);
    const v8h* s2 = (const v8h*)(Vt + (size_t)c * 12288);
    v8h* d0 = (v8h*)Qs; v8h* d1 = (v8h*)Ks; v8h* d2 = (v8h*)Vs;
#if HAVE_ASYNC_LDS
    for (int i = tid; i < 1536; i += 256) {
      async_copy_b128(s0 + i, d0 + i);
      async_copy_b128(s1 + i, d1 + i);
      async_copy_b128(s2 + i, d2 + i);
    }
    if (tid < 96) ctx[tid] = 0.f;
    async_wait0();
#else
    for (int i = tid; i < 1536; i += 256) { d0[i] = s0[i]; d1[i] = s1[i]; d2[i] = s2[i]; }
    if (tid < 96) ctx[tid] = 0.f;
#endif
  }
  __syncthreads();

  // --- scores: wave w owns q rows [16w,16w+16), 8 column tiles, K=96 ---
  const float scale = 0.1020620726159658f; // 1/sqrt(96)
  v8f sc[8];
#pragma unroll
  for (int t = 0; t < 8; ++t) sc[t] = v8f{};
#pragma unroll
  for (int ch = 0; ch < 3; ++ch) {
    v16h a = loadA(Qs, 96, w * 16, ch * 32, lane);
#pragma unroll
    for (int t = 0; t < 8; ++t) {
      v16h bb = loadB(Ks, 96, t * 16, ch * 32, lane);
      sc[t] = wmma16(a, bb, sc[t]);
    }
  }

  // --- in-register softmax over k (128 values split across 8 tiles x 16 lanes)
  const int nb = lane & 15;
#pragma unroll
  for (int r = 0; r < 8; ++r) {
    float m = -1e30f;
#pragma unroll
    for (int t = 0; t < 8; ++t) { float v = sc[t][r] * scale; sc[t][r] = v; m = fmaxf(m, v); }
    for (int off = 8; off >= 1; off >>= 1) m = fmaxf(m, __shfl_xor(m, off, 16));
    float s = 0.f;
#pragma unroll
    for (int t = 0; t < 8; ++t) { float e = __expf(sc[t][r] - m); sc[t][r] = e; s += e; }
    for (int off = 8; off >= 1; off >>= 1) s += __shfl_xor(s, off, 16);
    float rinv = 1.f / s;
    int q = w * 16 + ((lane < 16) ? r : r + 8);
#pragma unroll
    for (int t = 0; t < 8; ++t) As[(size_t)q * 128 + t * 16 + nb] = (_Float16)(sc[t][r] * rinv);
  }
  __syncthreads();

  // --- context: ctx[q][h] = attn[q][:] . V[:][h], then sum over q ---
  v8f cx[6];
#pragma unroll
  for (int t = 0; t < 6; ++t) cx[t] = v8f{};
#pragma unroll
  for (int ch = 0; ch < 4; ++ch) {
    v16h a = loadA(As, 128, w * 16, ch * 32, lane);
#pragma unroll
    for (int t = 0; t < 6; ++t) {
      v16h bb = loadB(Vs, 128, t * 16, ch * 32, lane);
      cx[t] = wmma16(a, bb, cx[t]);
    }
  }
#pragma unroll
  for (int t = 0; t < 6; ++t) {
    float s = 0.f;
#pragma unroll
    for (int r = 0; r < 8; ++r) s += cx[t][r];   // rows 0-7 (lo half) / 8-15 (hi half)
    s += __shfl_xor(s, 16, 32);                  // combine halves -> 16-row sum
    if (lane < 16) atomicAdd(&ctx[t * 16 + lane], s);
  }
  __syncthreads();

  if (tid == 0) {
    float mu = 0.f;
    for (int i = 0; i < 96; ++i) { float h = ctx[i] * (1.f / 128.f); ctx[i] = h; mu += h; }
    mu *= (1.f / 96.f);
    float var = 0.f;
    for (int i = 0; i < 96; ++i) { float d2 = ctx[i] - mu; var += d2 * d2; }
    var *= (1.f / 96.f);
    float rstd = rsqrtf(var + 1e-5f);
    float ss = 0.f;
    for (int i = 0; i < 96; ++i) {
      float h = (ctx[i] - mu) * rstd * ln_g[i] + ln_b[i];
      ctx[i] = h; ss += h * h;
    }
    float inv = 1.f / fmaxf(sqrtf(ss), 1e-6f);
    const float* dv = Pd + (size_t)b * 96;
    float dot = 0.f;
    for (int i = 0; i < 96; ++i) dot += dv[i] * ctx[i] * inv;
    S_TGL[b * 64 + c] = dot;
  }
}

// ---------------------------------------------------------------------------
// Kernel 3: S_CLL[b][c] = max over (t,p) of t_tok[b,t,:].v_pat[c,p,:]
//   128x196 sim per pair via WMMA (196 padded to 13 tiles with p<196 mask).
//   LDS (dynamic): Ts 24576 | Ps 39936 | red 32 = 64544 B
// ---------------------------------------------------------------------------
#define CLL_LDS_BYTES (24576 + 208 * 96 * 2 + 32)
__global__ __launch_bounds__(256)
void cll_kernel(const _Float16* __restrict__ Ttok,
                const _Float16* __restrict__ Vpat,
                float* __restrict__ S_CLL) {
  extern __shared__ char smem[];
  _Float16* Ts  = (_Float16*)(smem);            // [128][96]
  _Float16* Ps  = (_Float16*)(smem + 24576);    // [208][96]
  float*    red = (float*)(smem + 24576 + 208 * 96 * 2);

  const int b = blockIdx.x, c = blockIdx.y;
  const int tid = threadIdx.x, lane = tid & 31, w = tid >> 5;

  {
    const v8h* s0 = (const v8h*)(Ttok + (size_t)b * 12288);
    const v8h* s1 = (const v8h*)(Vpat + (size_t)c * 18816); // 196*96
    v8h* d0 = (v8h*)Ts;
    v8h* d1 = (v8h*)Ps;
#if HAVE_ASYNC_LDS
    for (int i = tid; i < 1536; i += 256) async_copy_b128(s0 + i, d0 + i);
    for (int i = tid; i < 2352; i += 256) async_copy_b128(s1 + i, d1 + i);
    for (int i = 18816 + tid; i < 208 * 96; i += 256) Ps[i] = (_Float16)0.f;
    async_wait0();
#else
    for (int i = tid; i < 1536; i += 256) d0[i] = s0[i];
    for (int i = tid; i < 2352; i += 256) d1[i] = s1[i];
    for (int i = 18816 + tid; i < 208 * 96; i += 256) Ps[i] = (_Float16)0.f;
#endif
  }
  __syncthreads();

  float vmax = -1e30f;
  const int pb = lane & 15;
  for (int tc = 0; tc < 13; ++tc) {
    v8f acc = v8f{};
#pragma unroll
    for (int ch = 0; ch < 3; ++ch) {
      v16h a  = loadA(Ts, 96, w * 16, ch * 32, lane);
      v16h bb = loadB(Ps, 96, tc * 16, ch * 32, lane);
      acc = wmma16(a, bb, acc);
    }
    if (tc * 16 + pb < 196) {
#pragma unroll
      for (int r = 0; r < 8; ++r) vmax = fmaxf(vmax, acc[r]);
    }
  }
  for (int off = 16; off >= 1; off >>= 1) vmax = fmaxf(vmax, __shfl_xor(vmax, off, 32));
  if (lane == 0) red[w] = vmax;
  __syncthreads();
  if (tid == 0) {
    float m = red[0];
    for (int i = 1; i < 8; ++i) m = fmaxf(m, red[i]);
    S_CLL[b * 64 + c] = m;
  }
}

// ---------------------------------------------------------------------------
// Kernel 4: out[i][j] = 0.5*(t[i].d[j] + S_TGL[i][j]) + 0.5*(t[i].v[j] + S_CLL[i][j])
// ---------------------------------------------------------------------------
__global__ __launch_bounds__(256)
void combine_kernel(const float* __restrict__ Pt, const float* __restrict__ Pd,
                    const float* __restrict__ Pv, const float* __restrict__ S_TGL,
                    const float* __restrict__ S_CLL, float* __restrict__ out) {
  int idx = blockIdx.x * 256 + threadIdx.x;
  if (idx >= 64 * 64) return;
  int i = idx >> 6, j = idx & 63;
  const float* t = Pt + (size_t)i * 96;
  const float* d = Pd + (size_t)j * 96;
  const float* v = Pv + (size_t)j * 96;
  float tgg = 0.f, cgg = 0.f;
  for (int h = 0; h < 96; ++h) { tgg += t[h] * d[h]; cgg += t[h] * v[h]; }
  out[idx] = 0.5f * (tgg + S_TGL[idx]) + 0.5f * (cgg + S_CLL[idx]);
}

// ---------------------------------------------------------------------------
extern "C" void kernel_launch(void* const* d_in, const int* in_sizes, int n_in,
                              void* d_out, int out_size, void* d_ws, size_t ws_size,
                              hipStream_t stream) {
  const float* z_d_cls   = (const float*)d_in[0];
  const float* Z_d_tok   = (const float*)d_in[1];
  const float* z_t_cls   = (const float*)d_in[2];
  const float* Z_t_tok   = (const float*)d_in[3];
  const float* z_v_cls   = (const float*)d_in[4];
  const float* Z_v_patch = (const float*)d_in[5];
  const float* W_t_cls   = (const float*)d_in[6];
  const float* b_t_cls   = (const float*)d_in[7];
  const float* W_d_cls   = (const float*)d_in[8];
  const float* b_d_cls   = (const float*)d_in[9];
  const float* W_v_cls   = (const float*)d_in[10];
  const float* b_v_cls   = (const float*)d_in[11];
  const float* W_t_tok   = (const float*)d_in[12];
  const float* b_t_tok   = (const float*)d_in[13];
  const float* W_v_patch = (const float*)d_in[14];
  const float* b_v_patch = (const float*)d_in[15];
  const float* W_tgl_q   = (const float*)d_in[16];
  const float* b_tgl_q   = (const float*)d_in[17];
  const float* W_tgl_k   = (const float*)d_in[18];
  const float* b_tgl_k   = (const float*)d_in[19];
  const float* W_tgl_v   = (const float*)d_in[20];
  const float* b_tgl_v   = (const float*)d_in[21];
  const float* ln_g      = (const float*)d_in[22];
  const float* ln_b      = (const float*)d_in[23];

  // workspace carve-up (~8.8 MB)
  char* ws = (char*)d_ws;
  size_t off = 0;
  auto carve = [&](size_t bytes) -> char* {
    char* p = ws + off;
    off += (bytes + 255) & ~(size_t)255;
    return p;
  };
  float*    Pt   = (float*)carve(64 * 96 * sizeof(float));
  float*    Pd   = (float*)carve(64 * 96 * sizeof(float));
  float*    Pv   = (float*)carve(64 * 96 * sizeof(float));
  _Float16* Qh   = (_Float16*)carve((size_t)64 * 128 * 96 * 2);
  _Float16* Kh   = (_Float16*)carve((size_t)64 * 128 * 96 * 2);
  _Float16* Vth  = (_Float16*)carve((size_t)64 * 96 * 128 * 2);
  _Float16* Tt   = (_Float16*)carve((size_t)64 * 128 * 96 * 2);
  _Float16* Vp   = (_Float16*)carve((size_t)64 * 196 * 96 * 2);
  float*    Stgl = (float*)carve(64 * 64 * sizeof(float));
  float*    Scll = (float*)carve(64 * 64 * sizeof(float));

  // Stage 1: projections (WMMA). cls -> f32 normalized; tokens -> f16.
  proj_kernel<<<1,  256, 0, stream>>>(z_t_cls,   64,    W_t_cls,   b_t_cls,   1, 0, nullptr, Pt);
  proj_kernel<<<1,  256, 0, stream>>>(z_d_cls,   64,    W_d_cls,   b_d_cls,   1, 0, nullptr, Pd);
  proj_kernel<<<1,  256, 0, stream>>>(z_v_cls,   64,    W_v_cls,   b_v_cls,   1, 0, nullptr, Pv);
  proj_kernel<<<64, 256, 0, stream>>>(Z_d_tok,   8192,  W_tgl_q,   b_tgl_q,   0, 0, Qh,  nullptr);
  proj_kernel<<<64, 256, 0, stream>>>(Z_t_tok,   8192,  W_tgl_k,   b_tgl_k,   0, 0, Kh,  nullptr);
  proj_kernel<<<64, 256, 0, stream>>>(Z_t_tok,   8192,  W_tgl_v,   b_tgl_v,   0, 1, Vth, nullptr);
  proj_kernel<<<64, 256, 0, stream>>>(Z_t_tok,   8192,  W_t_tok,   b_t_tok,   1, 0, Tt,  nullptr);
  proj_kernel<<<98, 256, 0, stream>>>(Z_v_patch, 12544, W_v_patch, b_v_patch, 1, 0, Vp,  nullptr);

  // Stage 2: cross attention per (b,c) pair -> S_TGL
  attn_kernel<<<dim3(64, 64), 256, ATTN_LDS_BYTES, stream>>>(Qh, Kh, Vth, Pd, ln_g, ln_b, Stgl);

  // Stage 3: token-patch max similarity -> S_CLL
  cll_kernel<<<dim3(64, 64), 256, CLL_LDS_BYTES, stream>>>(Tt, Vp, Scll);

  // Stage 4: combine all four similarity terms
  combine_kernel<<<16, 256, 0, stream>>>(Pt, Pd, Pv, Stgl, Scll, (float*)d_out);
}
